// NeuralAstar_83640193123018
// MI455X (gfx1250) — matrix operand order repaired and checked by reading, not compiled, not run on hardware
//
#include <hip/hip_runtime.h>
#include <hip/hip_bf16.h>
#include <math.h>

#define BATCH 8
#define HDIM 64
#define WDIM 64
#define NPIX (HDIM * WDIM)      // 4096
#define MPIX (BATCH * NPIX)     // 32768 total output pixels per layer
#define TSTEPS 1024             // NPIX * TMAX
#define LN_EPS 1e-6f
#define TBF 0.001f

typedef __attribute__((ext_vector_type(16))) __bf16 v16bf;
typedef __attribute__((ext_vector_type(16))) unsigned short v16u;
typedef __attribute__((ext_vector_type(8))) unsigned int v8u;
typedef __attribute__((ext_vector_type(8))) float v8f;

__device__ __forceinline__ unsigned short f2bf(float f) {
  unsigned int u = __float_as_uint(f);
  return (unsigned short)((u + 0x7FFFu + ((u >> 16) & 1u)) >> 16); // RNE
}

// ---------------------------------------------------------------------------
// Build x0 = stack(map, start+goal) in NHWC (C=2), bf16
// ---------------------------------------------------------------------------
__global__ void k_build_input(const float* __restrict__ map,
                              const float* __restrict__ start,
                              const float* __restrict__ goal,
                              unsigned short* __restrict__ x0) {
  int p = blockIdx.x * blockDim.x + threadIdx.x;
  if (p < MPIX) {
    x0[p * 2 + 0] = f2bf(map[p]);
    x0[p * 2 + 1] = f2bf(start[p] + goal[p]);
  }
}

// ---------------------------------------------------------------------------
// Pack conv weights [K x Cout] f32 into B-fragment-layout bf16:
//   packed[((kt*nTiles + nt)*32 + lane)*16 + j] = W[kt*32 + (lane>>4)*8 + (j<8 ? j : 8+j),
//                                                   nt*16 + (lane&15)]
// so each lane's whole 32B B fragment is one contiguous load.
// ---------------------------------------------------------------------------
__global__ void k_pack_w(const float* __restrict__ wgt,
                         unsigned short* __restrict__ packed,
                         int K, int Cout, int kTiles, int nTiles) {
  int idx = blockIdx.x * blockDim.x + threadIdx.x;
  int total = kTiles * nTiles * 32;
  if (idx >= total) return;
  const int lane = idx & 31;
  const int nt = (idx >> 5) % nTiles;
  const int kt = (idx >> 5) / nTiles;
  const int n = nt * 16 + (lane & 15);
  const int kbase = kt * 32 + (lane >> 4) * 8;
  v16u tmp;
#pragma unroll
  for (int j = 0; j < 16; ++j) {
    const int k = kbase + (j < 8 ? j : 8 + j);
    tmp[j] = (k < K && n < Cout) ? f2bf(wgt[k * Cout + n]) : (unsigned short)0;
  }
  ((v16u*)packed)[idx] = tmp;
}

// ---------------------------------------------------------------------------
// Implicit-GEMM conv3x3 SAME via WMMA bf16 -> f32, templated on shapes.
//   Activations are bf16 NHWC; A fragment 8-runs are single 16B loads.
//   M = 32768 pixels, K = 9*CIN, N = COUT. 4 waves/block; each wave owns a
//   16 x (16*NT) output tile: one A fragment feeds NT WMMAs per K-step.
// ---------------------------------------------------------------------------
template <int CIN, int COUT, int NT>
__global__ __launch_bounds__(128) void k_conv_wmma_t(
    const unsigned short* __restrict__ act,
    const unsigned short* __restrict__ wpk,
    const float* __restrict__ bias, float* __restrict__ out) {
  constexpr int K = 9 * CIN;
  constexpr int KT = (K + 31) / 32;
  constexpr int NTILES = (COUT + 15) / 16;

  const int wave = threadIdx.x >> 5;
  const int lane = threadIdx.x & 31;
  const int mTile = blockIdx.x * 4 + wave;     // MPIX/16 = 2048 tiles
  const int ntBase = blockIdx.y * NT;
  const int half = lane >> 4;
  const int p = mTile * 16 + (lane & 15);
  const int b = p >> 12;                        // NPIX = 4096
  const int rem = p & 4095;
  const int py = rem >> 6;
  const int px = rem & 63;

  v8f acc[NT];
#pragma unroll
  for (int q = 0; q < NT; ++q) acc[q] = (v8f){};

  const v16u* __restrict__ wv = (const v16u*)wpk;

  for (int kt = 0; kt < KT; ++kt) {
    v16bf afrag;
    if constexpr (CIN % 8 == 0) {
      // K multiple of 32 here; each 8-run of K is channel-contiguous bf16:
      // one 16-byte load per run, no conversion.
      v8u aw;
#pragma unroll
      for (int r = 0; r < 2; ++r) {
        const int kstart = kt * 32 + r * 16 + half * 8;
        const int t = kstart / CIN;             // compile-time divisor
        const int c0 = kstart - t * CIN;
        const int t3 = t / 3;
        const int dy = t3 - 1, dx = t - t3 * 3 - 1;
        const int yy = py + dy, xx = px + dx;
        uint4 u = {0u, 0u, 0u, 0u};
        if ((unsigned)yy < (unsigned)HDIM && (unsigned)xx < (unsigned)WDIM)
          u = *(const uint4*)(act + ((b * HDIM + yy) * WDIM + xx) * CIN + c0);
        aw[r * 4 + 0] = u.x; aw[r * 4 + 1] = u.y;
        aw[r * 4 + 2] = u.z; aw[r * 4 + 3] = u.w;
      }
      afrag = __builtin_bit_cast(v16bf, aw);
    } else {
      // generic bf16 gather (layer 0: CIN=2, K=18, single K-chunk)
      v16u au;
#pragma unroll
      for (int j = 0; j < 16; ++j) {
        const int v = j >> 1, par = j & 1;
        const int kk = (v < 4) ? (half * 8 + 2 * v + par)
                               : (16 + half * 8 + 2 * (v - 4) + par);
        const int ka = kt * 32 + kk;
        unsigned short aval = 0;
        if (ka < K) {
          const int t = ka / CIN;
          const int c = ka - t * CIN;
          const int t3 = t / 3;
          const int dy = t3 - 1, dx = t - t3 * 3 - 1;
          const int yy = py + dy, xx = px + dx;
          if ((unsigned)yy < (unsigned)HDIM && (unsigned)xx < (unsigned)WDIM)
            aval = act[((b * HDIM + yy) * WDIM + xx) * CIN + c];
        }
        au[j] = aval;
      }
      afrag = __builtin_bit_cast(v16bf, au);
    }
#pragma unroll
    for (int q = 0; q < NT; ++q) {
      const int nt = ntBase + q;
      const v16u bu = wv[(kt * NTILES + nt) * 32 + lane];
      const v16bf bfrag = __builtin_bit_cast(v16bf, bu);
      acc[q] = __builtin_amdgcn_wmma_f32_16x16x32_bf16(
          false, afrag, false, bfrag, (short)0, acc[q], false, false);
    }
  }

#pragma unroll
  for (int q = 0; q < NT; ++q) {
    const int ncol = (ntBase + q) * 16 + (lane & 15);
    if (ncol < COUT) {
      const float bv = bias[ncol];
#pragma unroll
      for (int r = 0; r < 8; ++r) {
        const int pp = mTile * 16 + r + 8 * half;
        out[pp * COUT + ncol] = acc[q][r] + bv;
      }
    }
  }
}

// ---------------------------------------------------------------------------
// LayerNorm (over channels) + ReLU (actmode=0) / sigmoid (actmode=1).
// Reads f32 conv output, writes bf16 activations (and optional f32 costmap).
// ---------------------------------------------------------------------------
__global__ void k_ln_act(const float* __restrict__ x,
                         unsigned short* __restrict__ dst,
                         const float* __restrict__ scale,
                         const float* __restrict__ lnb, int C, int actmode,
                         float* __restrict__ costOut) {
  int p = blockIdx.x * blockDim.x + threadIdx.x;
  if (p >= MPIX) return;
  float mu = 0.f;
  for (int c = 0; c < C; ++c) mu += x[p * C + c];
  mu /= (float)C;
  float var = 0.f;
  for (int c = 0; c < C; ++c) { float d = x[p * C + c] - mu; var += d * d; }
  var /= (float)C;
  const float rstd = rsqrtf(var + LN_EPS);
  for (int c = 0; c < C; ++c) {
    float v = (x[p * C + c] - mu) * rstd * scale[c] + lnb[c];
    v = actmode ? (1.f / (1.f + __expf(-v))) : fmaxf(v, 0.f);
    dst[p * C + c] = f2bf(v);
    if (costOut) costOut[p * C + c] = v;
  }
}

// ---------------------------------------------------------------------------
// h = chebyshev + tie-break euclid + cost  (one block per batch sample)
// ---------------------------------------------------------------------------
__global__ void k_heuristic(const float* __restrict__ goal,
                            const float* __restrict__ cost,
                            float* __restrict__ hout) {
  __shared__ int sgy, sgx;
  const int bi = blockIdx.x;
  const float* g = goal + bi * NPIX;
  for (int j = threadIdx.x; j < NPIX; j += blockDim.x)
    if (g[j] > 0.5f) { sgy = j / WDIM; sgx = j % WDIM; }
  __syncthreads();
  const float gy = (float)sgy, gx = (float)sgx;
  for (int j = threadIdx.x; j < NPIX; j += blockDim.x) {
    const float dy = fabsf((float)(j / WDIM) - gy);
    const float dx = fabsf((float)(j % WDIM) - gx);
    const float cheb = dy + dx - fminf(dy, dx);
    hout[bi * NPIX + j] = cheb + TBF * sqrtf(dy * dy + dx * dx)
                          + cost[bi * NPIX + j];
  }
}

// ---------------------------------------------------------------------------
// Persistent A* : one workgroup per batch sample. 256 thr, 16 cells/thread.
// Argmax = wave32 shfl_xor butterfly + tiny cross-wave merge (2 barriers).
// ---------------------------------------------------------------------------
__global__ __launch_bounds__(256) void k_astar(
    const float* __restrict__ cost, const float* __restrict__ hbuf,
    const float* __restrict__ obst, const float* __restrict__ start,
    const float* __restrict__ goal, float* __restrict__ gws,
    float* __restrict__ ows, float* __restrict__ hws, int* __restrict__ pws,
    float* __restrict__ out_hist, float* __restrict__ out_path) {
  const int bi = blockIdx.x;
  const int tid = threadIdx.x;
  const float* c_ = cost + bi * NPIX;
  const float* h_ = hbuf + bi * NPIX;
  const float* ob = obst + bi * NPIX;
  float* g_ = gws + bi * NPIX;
  float* op = ows + bi * NPIX;
  float* hi = hws + bi * NPIX;
  int*   pa = pws + bi * NPIX;

  __shared__ float sval[8];
  __shared__ int   sidx[8];
  __shared__ int s_sel, s_goal, s_start, s_solved;
  __shared__ float s_gc;

  if (tid == 0) s_solved = 0;
  for (int j = tid; j < NPIX; j += 256) {
    if (start[bi * NPIX + j] > 0.5f) s_start = j;
    if (goal[bi * NPIX + j] > 0.5f)  s_goal = j;
  }
  __syncthreads();
  const int startIdx = s_start;
  for (int j = tid; j < NPIX; j += 256) {
    g_[j] = 0.f; hi[j] = 0.f;
    op[j] = start[bi * NPIX + j];
    pa[j] = startIdx;
  }
  __syncthreads();

  const float inv_sqrt_n = 1.0f / 64.0f;  // 1/sqrt(4096)
  for (int step = 0; step < TSTEPS; ++step) {
    float bestv = -1.0f;
    int besti = tid * 16;
#pragma unroll
    for (int q = 0; q < 16; ++q) {
      const int j = tid * 16 + q;
      const float f = 0.5f * (g_[j] + h_[j]);      // G_RATIO = 0.5
      const float val = __expf(-f * inv_sqrt_n) * op[j];
      if (val > bestv) { bestv = val; besti = j; } // first-index tie-break
    }
    // wave32 butterfly reduction (ds_swizzle/permute, no barriers)
#pragma unroll
    for (int m = 16; m >= 1; m >>= 1) {
      const float v2 = __shfl_xor(bestv, m);
      const int   i2 = __shfl_xor(besti, m);
      if (v2 > bestv || (v2 == bestv && i2 < besti)) { bestv = v2; besti = i2; }
    }
    if ((tid & 31) == 0) { sval[tid >> 5] = bestv; sidx[tid >> 5] = besti; }
    __syncthreads();
    if (tid == 0) {
      float bv = sval[0]; int bix = sidx[0];
#pragma unroll
      for (int w = 1; w < 8; ++w)
        if (sval[w] > bv || (sval[w] == bv && sidx[w] < bix)) {
          bv = sval[w]; bix = sidx[w];
        }
      const int sel = bix;
      s_sel = sel;
      s_gc = g_[sel] + c_[sel];
      hi[sel] = 1.0f;           // hist += sel
      op[sel] = 0.0f;           // open -= sel
      if (sel == s_goal) s_solved = 1;
    }
    __syncthreads();
    const int sel = s_sel;
    if (tid < 9) {              // sparse 3x3 relaxation
      const int dy = tid / 3 - 1, dx = tid % 3 - 1;
      const int y = sel / WDIM + dy, x = sel % WDIM + dx;
      if ((unsigned)y < (unsigned)HDIM && (unsigned)x < (unsigned)WDIM) {
        const int n = y * WDIM + x;
        if (ob[n] > 0.5f && op[n] == 0.f && hi[n] == 0.f) {
          g_[n] = s_gc; op[n] = 1.0f; pa[n] = sel;
        }
      }
    }
    __syncthreads();
    if (s_solved) break;        // == reference 'active' freeze
  }

  for (int j = tid; j < NPIX; j += 256) {
    out_hist[bi * NPIX + j] = hi[j];
    out_path[bi * NPIX + j] = 0.f;
  }
  __syncthreads();
  if (tid == 0) {               // backtrack (fixed point at start cell)
    const int gl = s_goal;
    out_path[bi * NPIX + gl] = 1.0f;
    int loc = pa[gl];
    for (int t = 0; t < TSTEPS; ++t) {
      out_path[bi * NPIX + loc] = 1.0f;
      const int nxt = pa[loc];
      if (nxt == loc) break;
      loc = nxt;
    }
  }
}

// ---------------------------------------------------------------------------
extern "C" void kernel_launch(void* const* d_in, const int* in_sizes, int n_in,
                              void* d_out, int out_size, void* d_ws,
                              size_t ws_size, hipStream_t stream) {
  (void)in_sizes; (void)n_in; (void)out_size; (void)ws_size;
  const float* map   = (const float*)d_in[0];
  const float* start = (const float*)d_in[1];
  const float* goal  = (const float*)d_in[2];
  const float *Wk[5], *Bc[5], *Sc[5], *Lb[5];
  for (int i = 0; i < 5; ++i) {
    Wk[i] = (const float*)d_in[3 + i];
    Bc[i] = (const float*)d_in[8 + i];
    Sc[i] = (const float*)d_in[13 + i];
    Lb[i] = (const float*)d_in[18 + i];
  }
  float* out_hist = (float*)d_out;
  float* out_path = (float*)d_out + MPIX;

  char* ws = (char*)d_ws;
  size_t off = 0;
  auto wsalloc = [&](size_t bytes) -> void* {
    void* p = ws + off;
    off += (bytes + 255) & ~(size_t)255;
    return p;
  };
  // bf16 activation ping-pong + shared f32 conv output
  unsigned short* bfA = (unsigned short*)wsalloc((size_t)MPIX * 256 * 2); // C=2,64,256
  unsigned short* bfB = (unsigned short*)wsalloc((size_t)MPIX * 128 * 2); // C=32,128,1
  float* convF = (float*)wsalloc((size_t)MPIX * 256 * sizeof(float));
  float* cmap  = (float*)wsalloc((size_t)MPIX * sizeof(float));
  float* hbuf  = (float*)wsalloc((size_t)MPIX * sizeof(float));
  float* gbuf  = (float*)wsalloc((size_t)MPIX * sizeof(float));
  float* obuf  = (float*)wsalloc((size_t)MPIX * sizeof(float));
  float* hibuf = (float*)wsalloc((size_t)MPIX * sizeof(float));
  int*   pbuf  = (int*)wsalloc((size_t)MPIX * sizeof(int));

  // per-layer packed weights: kTiles*nTiles*32 fragments * 32B
  const int KTS[5] = {1, 9, 18, 36, 72};
  const int NTS[5] = {2, 4, 8, 16, 1};
  const int KS[5]  = {18, 288, 576, 1152, 2304};
  const int CO[5]  = {32, 64, 128, 256, 1};
  unsigned short* wp[5];
  for (int i = 0; i < 5; ++i)
    wp[i] = (unsigned short*)wsalloc((size_t)KTS[i] * NTS[i] * 32 * 16 *
                                     sizeof(unsigned short));

  for (int i = 0; i < 5; ++i) {
    const int total = KTS[i] * NTS[i] * 32;
    k_pack_w<<<(total + 255) / 256, 256, 0, stream>>>(Wk[i], wp[i], KS[i],
                                                      CO[i], KTS[i], NTS[i]);
  }

  k_build_input<<<(MPIX + 255) / 256, 256, 0, stream>>>(map, start, goal, bfA);

  const int MG = MPIX / 16 / 4;  // 512 blocks along M
  const int PB = (MPIX + 255) / 256;
  // layer 0: 2 -> 32  (NT=2)
  k_conv_wmma_t<2, 32, 2><<<dim3(MG, 1), 128, 0, stream>>>(bfA, wp[0], Bc[0], convF);
  k_ln_act<<<PB, 256, 0, stream>>>(convF, bfB, Sc[0], Lb[0], 32, 0, nullptr);
  // layer 1: 32 -> 64 (NT=4)
  k_conv_wmma_t<32, 64, 4><<<dim3(MG, 1), 128, 0, stream>>>(bfB, wp[1], Bc[1], convF);
  k_ln_act<<<PB, 256, 0, stream>>>(convF, bfA, Sc[1], Lb[1], 64, 0, nullptr);
  // layer 2: 64 -> 128 (NT=4, grid.y=2)
  k_conv_wmma_t<64, 128, 4><<<dim3(MG, 2), 128, 0, stream>>>(bfA, wp[2], Bc[2], convF);
  k_ln_act<<<PB, 256, 0, stream>>>(convF, bfB, Sc[2], Lb[2], 128, 0, nullptr);
  // layer 3: 128 -> 256 (NT=4, grid.y=4)
  k_conv_wmma_t<128, 256, 4><<<dim3(MG, 4), 128, 0, stream>>>(bfB, wp[3], Bc[3], convF);
  k_ln_act<<<PB, 256, 0, stream>>>(convF, bfA, Sc[3], Lb[3], 256, 1, nullptr);
  // layer 4: 256 -> 1 (NT=1)
  k_conv_wmma_t<256, 1, 1><<<dim3(MG, 1), 128, 0, stream>>>(bfA, wp[4], Bc[4], convF);
  k_ln_act<<<PB, 256, 0, stream>>>(convF, bfB, Sc[4], Lb[4], 1, 1, cmap);

  k_heuristic<<<BATCH, 256, 0, stream>>>(goal, cmap, hbuf);
  k_astar<<<BATCH, 256, 0, stream>>>(cmap, hbuf, map, start, goal, gbuf,
                                     obuf, hibuf, pbuf, out_hist, out_path);
}
